// TopKPooling_14705968021522
// MI455X (gfx1250) — compile-verified, matrix operand order checked by reading, not run on hardware
//
#include <hip/hip_runtime.h>
#include <hip/hip_bf16.h>
#include <math.h>

// ---------------- problem constants (from reference) ----------------
constexpr int Bn     = 8;
constexpr int N_PER  = 2048;
constexpr int Ssent  = 64;
constexpr int IN_CH  = 256;
constexpr int HID    = 512;
constexpr int DEGC   = 16;
constexpr int Nn     = Bn * N_PER;        // 16384
constexpr int En     = Nn * DEGC;         // 262144
constexpr int OTH    = N_PER - Ssent;     // 1984
constexpr int K_OTH  = 992;               // ceil(0.5*1984)
constexpr int NOUT   = 3;
constexpr int Pn     = Bn * NOUT + Bn * K_OTH;  // 7960

// ---------------- output flat offsets (reference return order) ------
constexpr long OFF_XOUT = 0;                       // (Pn,256)
constexpr long OFF_EI   = (long)Pn * IN_CH;        // (2,En)
constexpr long OFF_EA   = OFF_EI + 2L * En;        // (En,1)
constexpr long OFF_BOUT = OFF_EA + En;             // (Pn,)
constexpr long OFF_PERM = OFF_BOUT + Pn;           // (Pn,)
constexpr long OFF_SC   = OFF_PERM + Pn;           // (Pn,)
constexpr long OFF_SS   = OFF_SC + Pn;             // (Bn*Ssent,)
constexpr long OFF_SB   = OFF_SS + (long)Bn * Ssent; // (Bn*Ssent,)

typedef __attribute__((ext_vector_type(16))) __bf16 v16bf;
typedef __attribute__((ext_vector_type(8)))  float  v8f;

__device__ __forceinline__ unsigned short f2bf(float f) {
  union { float f; unsigned u; } v; v.f = f;
  unsigned r = v.u + 0x7FFFu + ((v.u >> 16) & 1u);   // round-to-nearest-even
  return (unsigned short)(r >> 16);
}
__device__ __forceinline__ float lrelu(float x) { return x > 0.0f ? x : 0.01f * x; }

// =============== pack X (f32, row-major MxK) into WMMA A lane layout ===========
// layout: xp[((mt*8+kt)*32 + lane)*16 + slot]; lane<16 holds rows m=mt*16+lane,
// k in {0..7,16..23}+kt*32; lane>=16 same rows, k in {8..15,24..31}+kt*32.
__global__ void k_pack_x(const float* __restrict__ x, unsigned short* __restrict__ xp) {
  int t = blockIdx.x * blockDim.x + threadIdx.x;
  if (t >= Nn * IN_CH) return;
  int slot = t & 15;
  int lane = (t >> 4) & 31;
  int kt   = (t >> 9) & 7;
  int mt   = t >> 12;
  int m = mt * 16 + (lane & 15);
  int k = kt * 32 + ((lane >> 4) << 3) + (slot < 8 ? slot : slot + 8);
  xp[t] = f2bf(x[(size_t)m * IN_CH + k]);
}

// =============== pack W1 (f32, KxN row-major) into WMMA B lane layout ==========
__global__ void k_pack_w(const float* __restrict__ w1, unsigned short* __restrict__ wp) {
  int t = blockIdx.x * blockDim.x + threadIdx.x;
  if (t >= IN_CH * HID) return;
  int slot = t & 15;
  int lane = (t >> 4) & 31;
  int kt   = (t >> 9) & 7;
  int nt   = t >> 12;                    // 0..31
  int n = nt * 16 + (lane & 15);
  int k = kt * 32 + ((lane >> 4) << 3) + (slot < 8 ? slot : slot + 8);
  wp[t] = f2bf(w1[(size_t)k * HID + n]);
}

// =============== WMMA GEMM: HX(16384x512) = Xbf16 @ W1bf16, f32 accum ==========
// One wave -> one 16x16 C tile, K=256 as 8 steps of v_wmma_f32_16x16x32_bf16.
// Fragments are staged into LDS with GLOBAL_LOAD_ASYNC_TO_LDS_B128 (ASYNCcnt),
// double-buffered per wave: async copy of stage kt+1 overlaps ds_load+wmma of kt.
__global__ void k_gemm1(const unsigned short* __restrict__ xp,
                        const unsigned short* __restrict__ wp,
                        float* __restrict__ HX) {
  // per-wave slice: [buf(2)][ A:1024B | B:1024B ] = 4KB; 8 waves -> 32KB
  __shared__ __attribute__((aligned(32))) unsigned char smem[8 * 4096];
  int lane = threadIdx.x & 31;
  int wv   = threadIdx.x >> 5;                 // 0..7
  int wave = blockIdx.x * 8 + wv;
  int mt = wave >> 5;                          // 0..1023
  int nt = wave & 31;                          // 0..31

  unsigned ldsSlice = ((unsigned)(size_t)smem) + (unsigned)(wv * 4096);
  unsigned ldsLane  = ldsSlice + (unsigned)(lane * 32);

  unsigned long long aBase = (unsigned long long)(size_t)xp +
                             ((unsigned long long)(mt * 8) * 32ull + (unsigned)lane) * 32ull;
  unsigned long long bBase = (unsigned long long)(size_t)wp +
                             ((unsigned long long)(nt * 8) * 32ull + (unsigned)lane) * 32ull;

  auto issue = [&](int kt) {
    unsigned la = ldsLane + (unsigned)((kt & 1) * 2048);
    unsigned lb = la + 1024;
    unsigned long long ga = aBase + (unsigned long long)kt * 1024ull;  // 32 lanes * 32B
    unsigned long long gb = bBase + (unsigned long long)kt * 1024ull;
    asm volatile("global_load_async_to_lds_b128 %0, %1, off"
                 :: "v"(la), "v"(ga) : "memory");
    asm volatile("global_load_async_to_lds_b128 %0, %1, off offset:16"
                 :: "v"(la), "v"(ga) : "memory");
    asm volatile("global_load_async_to_lds_b128 %0, %1, off"
                 :: "v"(lb), "v"(gb) : "memory");
    asm volatile("global_load_async_to_lds_b128 %0, %1, off offset:16"
                 :: "v"(lb), "v"(gb) : "memory");
  };

  v8f acc = {};
  issue(0);
#pragma unroll
  for (int kt = 0; kt < 8; ++kt) {
    if (kt < 7) {
      issue(kt + 1);
      // 4 async ops of stage kt+1 in flight; async loads complete in order,
      // so <=4 outstanding guarantees stage kt's data landed in LDS.
      asm volatile("s_wait_asynccnt 4" ::: "memory");
    } else {
      asm volatile("s_wait_asynccnt 0" ::: "memory");
    }
    const unsigned char* bufp = smem + wv * 4096 + (kt & 1) * 2048 + lane * 32;
    v16bf a = *reinterpret_cast<const v16bf*>(bufp);
    v16bf b = *reinterpret_cast<const v16bf*>(bufp + 1024);
    acc = __builtin_amdgcn_wmma_f32_16x16x32_bf16(
        /*neg_a=*/false, a, /*neg_b=*/false, b,
        /*c_mod=*/(short)0, acc, /*reuse_a=*/false, /*reuse_b=*/false);
  }
  __builtin_amdgcn_s_wait_tensorcnt(0);   // gfx1250 split-counter wait (no-op here)
  int row0 = (lane >> 4) << 3;
  int col  = lane & 15;
#pragma unroll
  for (int r = 0; r < 8; ++r) {
    int m = mt * 16 + row0 + r;
    HX[(size_t)m * HID + nt * 16 + col] = acc[r];
  }
}

// =============== degree / norm ===============
__global__ void k_deg_init(float* __restrict__ deg) {
  int i = blockIdx.x * blockDim.x + threadIdx.x;
  if (i < Nn) deg[i] = 1.0f;   // self loop, weight 1
}
__global__ void k_deg_edges(const int* __restrict__ col, const float* __restrict__ ew,
                            float* __restrict__ deg) {
  int e = blockIdx.x * blockDim.x + threadIdx.x;
  if (e < En) atomicAdd(&deg[col[e]], ew[e]);
}
__global__ void k_dinv(const float* __restrict__ deg, float* __restrict__ dinv) {
  int i = blockIdx.x * blockDim.x + threadIdx.x;
  if (i < Nn) { float d = deg[i]; dinv[i] = d > 0.0f ? rsqrtf(d) : 0.0f; }
}

// =============== GCN1 aggregation ===============
__global__ void k_h_init(const float* __restrict__ HX, const float* __restrict__ dinv,
                         const float* __restrict__ b1, float* __restrict__ H) {
  long t = (long)blockIdx.x * blockDim.x + threadIdx.x;
  if (t >= (long)Nn * HID) return;
  int i = (int)(t >> 9);
  int f = (int)(t & (HID - 1));
  float di = dinv[i];
  H[t] = b1[f] + di * di * HX[t];
}
__global__ void k_h_edges(const int* __restrict__ row, const int* __restrict__ col,
                          const float* __restrict__ ew, const float* __restrict__ dinv,
                          const float* __restrict__ HX, float* __restrict__ H) {
  long t = (long)blockIdx.x * blockDim.x + threadIdx.x;   // En * 64 threads
  if (t >= (long)En * 64) return;
  int e  = (int)(t >> 6);
  int f0 = (int)(t & 63) << 3;
  int r = row[e], c = col[e];
  float nrm = dinv[r] * ew[e] * dinv[c];
  const float* src = HX + (size_t)r * HID + f0;
  float*       dst = H  + (size_t)c * HID + f0;
#pragma unroll
  for (int j = 0; j < 8; ++j) atomicAdd(&dst[j], nrm * src[j]);
}
__global__ void k_lrelu(float* __restrict__ H) {
  long t = (long)blockIdx.x * blockDim.x + threadIdx.x;
  if (t < (long)Nn * HID) H[t] = lrelu(H[t]);
}

// =============== GCN2: h @ W2 (512 -> 1) then scalar aggregation ===============
__global__ void k_slin(const float* __restrict__ H, const float* __restrict__ W2,
                       float* __restrict__ slin) {
  int i = blockIdx.x * blockDim.x + threadIdx.x;
  if (i >= Nn) return;
  const float4* h4 = reinterpret_cast<const float4*>(H + (size_t)i * HID);
  const float4* w4 = reinterpret_cast<const float4*>(W2);
  float s = 0.0f;
#pragma unroll 4
  for (int f = 0; f < HID / 4; ++f) {
    float4 a = h4[f], b = w4[f];
    s += a.x * b.x + a.y * b.y + a.z * b.z + a.w * b.w;
  }
  slin[i] = s;
}
__global__ void k_score_init(const float* __restrict__ slin, const float* __restrict__ dinv,
                             const float* __restrict__ b2, float* __restrict__ score) {
  int i = blockIdx.x * blockDim.x + threadIdx.x;
  if (i < Nn) { float di = dinv[i]; score[i] = b2[0] + di * di * slin[i]; }
}
__global__ void k_score_edges(const int* __restrict__ row, const int* __restrict__ col,
                              const float* __restrict__ ew, const float* __restrict__ dinv,
                              const float* __restrict__ slin, float* __restrict__ score) {
  int e = blockIdx.x * blockDim.x + threadIdx.x;
  if (e >= En) return;
  int r = row[e], c = col[e];
  atomicAdd(&score[c], dinv[r] * ew[e] * dinv[c] * slin[r]);
}
__global__ void k_nan2num(float* __restrict__ score) {
  int i = blockIdx.x * blockDim.x + threadIdx.x;
  if (i >= Nn) return;
  float v = score[i];
  if (isnan(v)) v = 0.0f;
  else if (isinf(v)) v = v > 0.0f ? 3.4028234663852886e38f : -3.4028234663852886e38f;
  score[i] = v;
}

// =============== stable top-k by rank counting (matches jax.lax.top_k order) ====
__global__ void k_topk_sent(const float* __restrict__ score, int* __restrict__ perm) {
  __shared__ float sv[Ssent];
  int b = blockIdx.x, i = threadIdx.x;
  sv[i] = score[b * N_PER + i];
  __syncthreads();
  float v = sv[i];
  int rank = 0;
  for (int j = 0; j < Ssent; ++j) {
    float vj = sv[j];
    rank += (vj > v) || (vj == v && j < i);
  }
  if (rank < NOUT) perm[b * NOUT + rank] = b * N_PER + i;
}
__global__ void k_topk_other(const float* __restrict__ score, int* __restrict__ perm) {
  __shared__ float sv[OTH];
  int b = blockIdx.x;
  for (int i = threadIdx.x; i < OTH; i += blockDim.x)
    sv[i] = score[b * N_PER + Ssent + i];
  __syncthreads();
  for (int i = threadIdx.x; i < OTH; i += blockDim.x) {
    float v = sv[i];
    int rank = 0;
    for (int j = 0; j < OTH; ++j) {
      float vj = sv[j];
      rank += (vj > v) || (vj == v && j < i);
    }
    if (rank < K_OTH) perm[Bn * NOUT + b * K_OTH + rank] = b * N_PER + Ssent + i;
  }
}

// =============== mask / remap / outputs ===============
__global__ void k_mask_init(int* __restrict__ mask) {
  int i = blockIdx.x * blockDim.x + threadIdx.x;
  if (i < Nn) mask[i] = -1;
}
__global__ void k_mask_set(const int* __restrict__ perm, int* __restrict__ mask) {
  int j = blockIdx.x * blockDim.x + threadIdx.x;
  if (j < Pn) mask[perm[j]] = j;
}
__global__ void k_xout(const float* __restrict__ x, const float* __restrict__ score,
                       const int* __restrict__ perm, float* __restrict__ out) {
  long t = (long)blockIdx.x * blockDim.x + threadIdx.x;
  if (t >= (long)Pn * IN_CH) return;
  int j = (int)(t >> 8);
  int f = (int)(t & (IN_CH - 1));
  int node = perm[j];
  out[OFF_XOUT + t] = x[(size_t)node * IN_CH + f] * score[node];
}
__global__ void k_edges_out(const int* __restrict__ row, const int* __restrict__ col,
                            const float* __restrict__ ea, const int* __restrict__ mask,
                            float* __restrict__ out) {
  int e = blockIdx.x * blockDim.x + threadIdx.x;
  if (e >= En) return;
  int rn = mask[row[e]], cn = mask[col[e]];
  bool valid = (rn >= 0) && (cn >= 0);
  out[OFF_EI + e]      = (float)(valid ? rn : -1);
  out[OFF_EI + En + e] = (float)(valid ? cn : -1);
  out[OFF_EA + e]      = valid ? ea[e] : 0.0f;
}
__global__ void k_tail(const int* __restrict__ perm, const float* __restrict__ score,
                       float* __restrict__ out) {
  int j = blockIdx.x * blockDim.x + threadIdx.x;
  if (j >= Pn) return;
  int node = perm[j];
  out[OFF_BOUT + j] = (float)(node / N_PER);
  out[OFF_PERM + j] = (float)node;
  out[OFF_SC + j]   = score[node];
}
__global__ void k_sent_out(const float* __restrict__ score, float* __restrict__ out) {
  int t = blockIdx.x * blockDim.x + threadIdx.x;
  if (t >= Bn * Ssent) return;
  int b = t / Ssent, s = t % Ssent;
  out[OFF_SS + t] = score[b * N_PER + s];
  out[OFF_SB + t] = (float)b;
}

// =============== host side ===============
static inline int gridFor(long n, int blk) { return (int)((n + blk - 1) / blk); }

extern "C" void kernel_launch(void* const* d_in, const int* in_sizes, int n_in,
                              void* d_out, int out_size, void* d_ws, size_t ws_size,
                              hipStream_t stream) {
  const float* x    = (const float*)d_in[0];
  const int*   eidx = (const int*)d_in[1];
  const int*   row  = eidx;
  const int*   col  = eidx + En;
  const float* ea   = (const float*)d_in[2];
  const float* W1   = (const float*)d_in[6];
  const float* b1   = (const float*)d_in[7];
  const float* W2   = (const float*)d_in[8];
  const float* b2   = (const float*)d_in[9];
  float* out = (float*)d_out;

  // workspace carve-out (256B aligned regions)
  char* ws = (char*)d_ws;
  size_t off = 0;
  auto carve = [&](size_t bytes) { char* p = ws + off; off = (off + bytes + 255) & ~(size_t)255; return p; };
  unsigned short* xp   = (unsigned short*)carve((size_t)Nn * IN_CH * 2);
  unsigned short* wp   = (unsigned short*)carve((size_t)IN_CH * HID * 2);
  float* HX   = (float*)carve((size_t)Nn * HID * 4);
  float* H    = (float*)carve((size_t)Nn * HID * 4);
  float* deg  = (float*)carve((size_t)Nn * 4);
  float* dinv = (float*)carve((size_t)Nn * 4);
  float* slin = (float*)carve((size_t)Nn * 4);
  float* scr  = (float*)carve((size_t)Nn * 4);
  int*   perm = (int*)carve((size_t)Pn * 4);
  int*   mask = (int*)carve((size_t)Nn * 4);
  (void)ws_size; (void)in_sizes; (void)n_in; (void)out_size;

  const int BLK = 256;

  // pack operands into WMMA lane layout (f32 -> bf16 RNE)
  k_pack_x<<<gridFor((long)Nn * IN_CH, BLK), BLK, 0, stream>>>(x, xp);
  k_pack_w<<<gridFor((long)IN_CH * HID, BLK), BLK, 0, stream>>>(W1, wp);

  // GEMM1: 1024 x 32 tiles, 1 wave each, 8 waves/block, async-LDS double buffer
  k_gemm1<<<(1024 * 32) / 8, BLK, 0, stream>>>(xp, wp, HX);

  // degrees & normalization
  k_deg_init<<<gridFor(Nn, BLK), BLK, 0, stream>>>(deg);
  k_deg_edges<<<gridFor(En, BLK), BLK, 0, stream>>>(col, ea, deg);
  k_dinv<<<gridFor(Nn, BLK), BLK, 0, stream>>>(deg, dinv);

  // GCN1 aggregation + bias + leaky_relu
  k_h_init<<<gridFor((long)Nn * HID, BLK), BLK, 0, stream>>>(HX, dinv, b1, H);
  k_h_edges<<<gridFor((long)En * 64, BLK), BLK, 0, stream>>>(row, col, ea, dinv, HX, H);
  k_lrelu<<<gridFor((long)Nn * HID, BLK), BLK, 0, stream>>>(H);

  // GCN2 scoring
  k_slin<<<gridFor(Nn, BLK), BLK, 0, stream>>>(H, W2, slin);
  k_score_init<<<gridFor(Nn, BLK), BLK, 0, stream>>>(slin, dinv, b2, scr);
  k_score_edges<<<gridFor(En, BLK), BLK, 0, stream>>>(row, col, ea, dinv, slin, scr);
  k_nan2num<<<gridFor(Nn, BLK), BLK, 0, stream>>>(scr);

  // top-k -> perm
  k_topk_sent<<<Bn, Ssent, 0, stream>>>(scr, perm);
  k_topk_other<<<Bn, BLK, 0, stream>>>(scr, perm);

  // mask + outputs
  k_mask_init<<<gridFor(Nn, BLK), BLK, 0, stream>>>(mask);
  k_mask_set<<<gridFor(Pn, BLK), BLK, 0, stream>>>(perm, mask);
  k_xout<<<gridFor((long)Pn * IN_CH, BLK), BLK, 0, stream>>>(x, scr, perm, out);
  k_edges_out<<<gridFor(En, BLK), BLK, 0, stream>>>(row, col, ea, mask, out);
  k_tail<<<gridFor(Pn, BLK), BLK, 0, stream>>>(perm, scr, out);
  k_sent_out<<<gridFor(Bn * Ssent, BLK), BLK, 0, stream>>>(scr, out);
}
